// DeBiLevelRoutingAttentionblcok_31817117728881
// MI455X (gfx1250) — matrix operand order repaired and checked by reading, not compiled, not run on hardware
//
#include <hip/hip_runtime.h>
#include <hip/hip_bf16.h>
#include <cstdint>

// ---------- CDNA5 WMMA types ----------
typedef __attribute__((ext_vector_type(16))) _Float16 v16h;
typedef __attribute__((ext_vector_type(8)))  _Float16 v8h;
typedef __attribute__((ext_vector_type(4)))  _Float16 v4h;
typedef __attribute__((ext_vector_type(8)))  float    v8f;
typedef __attribute__((ext_vector_type(4)))  float    v4f;

#define BATCH 16
#define DIMC  512
#define OC    1536   // 2*QK + DIM
#define HH    56
#define WW    56
#define HWPIX 3136
#define NWIN  7
#define WH    8
#define P2    49
#define W2    64
#define HEADS 8
#define HDQ   64
#define TOPK  4

static __device__ __forceinline__ int win_pix(int p, int e) {
    // window p in [0,49), element e in [0,64) -> flat pixel in 56x56
    int wy = p / NWIN, wx = p - wy * NWIN;
    int y = e >> 3, x = e & 7;
    return (wy * WH + y) * WW + (wx * WH + x);
}

// A-fragment: lane half h holds K = h*16..h*16+15 (contiguous, matches frag_b)
// -> one contiguous 32-byte run of the row.
static __device__ __forceinline__ v16h frag_a(const _Float16* row, int half) {
    return *(const v16h*)(row + half * 16);
}
// B-fragment (contiguous K order, matches frag_a): lane holds 16 contiguous K values
// of one N column when the tile is stored N-major.
static __device__ __forceinline__ v16h frag_b(const _Float16* colk) {
    return *(const v16h*)colk;
}

// =====================================================================
// Kernel 1: QKV projection GEMM.  Per batch n:
//   qkv[n][o][p] = sum_c w[o][c] * x[n][c][p] + b[o]
// Block tile 128(O) x 32(P), 8 waves, each wave a 16x32 strip:
// one A-fragment feeds two v_wmma_f32_16x16x32_f16 per K-step.
// =====================================================================
__global__ __launch_bounds__(256)
void qkv_gemm_kernel(const float* __restrict__ x, const float* __restrict__ w,
                     const float* __restrict__ b, float* __restrict__ qkv) {
    __shared__ alignas(32) _Float16 As[128 * 32];   // w tile, rows=O cols=K
    __shared__ alignas(32) _Float16 Bst[32 * 32];   // x tile, p-major: [p][k]

    const int n  = blockIdx.z;
    const int ob = blockIdx.y * 128;
    const int pb = blockIdx.x * 32;
    const int tid  = threadIdx.x;
    const int lane = tid & 31;
    const int wave = tid >> 5;
    const int l15  = lane & 15;
    const int half = lane >> 4;

    const float* xb = x + (size_t)n * DIMC * HWPIX;

    v8f acc0 = {}, acc1 = {};
    for (int k0 = 0; k0 < DIMC; k0 += 32) {
        // stage A: 128x32 of w via float4 loads (row-major, stride DIMC)
        for (int i = tid; i < 1024; i += 256) {
            int r = i >> 3, c4 = (i & 7) << 2;
            v4f t = *(const v4f*)&w[(size_t)(ob + r) * DIMC + (k0 + c4)];
            v4h hh = { (_Float16)t[0], (_Float16)t[1], (_Float16)t[2], (_Float16)t[3] };
            *(v4h*)&As[r * 32 + c4] = hh;
        }
        // stage B: 32(K)x32(P) of x via float4 loads, stored p-major
        if (tid < 256) {
            int r = tid >> 3, p4 = (tid & 7) << 2;
            v4f t = *(const v4f*)&xb[(size_t)(k0 + r) * HWPIX + (pb + p4)];
#pragma unroll
            for (int j = 0; j < 4; ++j)
                Bst[(p4 + j) * 32 + r] = (_Float16)t[j];
        }
        if (k0 + 32 < DIMC)
            __builtin_prefetch(&xb[(size_t)(k0 + 32) * HWPIX + pb], 0, 1);
        __syncthreads();

        v16h a = frag_a(&As[(wave * 16 + l15) * 32], half);
        v16h b0 = frag_b(&Bst[l15 * 32 + half * 16]);
        v16h b1 = frag_b(&Bst[(16 + l15) * 32 + half * 16]);
        acc0 = __builtin_amdgcn_wmma_f32_16x16x32_f16(false, a, false, b0,
                                                      (short)0, acc0, false, false);
        acc1 = __builtin_amdgcn_wmma_f32_16x16x32_f16(false, a, false, b1,
                                                      (short)0, acc1, false, false);
        __syncthreads();
    }

#pragma unroll
    for (int v = 0; v < 8; ++v) {
        int o = ob + wave * 16 + half * 8 + v;
        float bias = b[o];
        qkv[((size_t)n * OC + o) * HWPIX + (pb + l15)]      = acc0[v] + bias;
        qkv[((size_t)n * OC + o) * HWPIX + (pb + 16 + l15)] = acc1[v] + bias;
    }
}

// =====================================================================
// Kernel 2: per-window means of q and k channels.
// =====================================================================
__global__ void winmean_kernel(const float* __restrict__ qkv,
                               float* __restrict__ qw, float* __restrict__ kw) {
    int idx = blockIdx.x * blockDim.x + threadIdx.x;
    if (idx >= BATCH * P2 * DIMC) return;
    int c = idx & (DIMC - 1);
    int p = (idx >> 9) % P2;
    int n = idx / (P2 * DIMC);
    const float* qp = qkv + ((size_t)n * OC + c) * HWPIX;
    const float* kp = qkv + ((size_t)n * OC + DIMC + c) * HWPIX;
    float sq = 0.f, sk = 0.f;
    for (int e = 0; e < W2; ++e) {
        int pix = win_pix(p, e);
        sq += qp[pix];
        sk += kp[pix];
    }
    qw[idx] = sq * (1.f / 64.f);
    kw[idx] = sk * (1.f / 64.f);
}

// =====================================================================
// Kernel 3: routing logits (49x49 per batch) + top-4 indices.
// Tie-break: lowest index first (matches jax.lax.top_k).
// =====================================================================
__global__ void routing_kernel(const float* __restrict__ qw, const float* __restrict__ kw,
                               int* __restrict__ ridx) {
    __shared__ float lg[P2];
    const int p = blockIdx.x, n = blockIdx.y, tid = threadIdx.x;
    if (tid < P2) {
        const float* q = qw + ((size_t)(n * P2) + p)   * DIMC;
        const float* k = kw + ((size_t)(n * P2) + tid) * DIMC;
        float s = 0.f;
        for (int c = 0; c < DIMC; ++c) s += q[c] * k[c];
        lg[tid] = s * 0.044194173824159216f;   // 512^-0.5
    }
    __syncthreads();
    if (tid == 0) {
        int sel[TOPK];
        for (int t = 0; t < TOPK; ++t) {
            float best = -3.4e38f; int bi = 0;
            for (int m = 0; m < P2; ++m) {
                bool used = false;
                for (int u = 0; u < t; ++u) used |= (sel[u] == m);
                if (!used && lg[m] > best) { best = lg[m]; bi = m; }
            }
            sel[t] = bi;
            ridx[((size_t)(n * P2) + p) * TOPK + t] = bi;
        }
    }
}

// =====================================================================
// Kernel 4: fused attention per (head, window, batch).
// Flash-style over the 4 routed key windows (chunks of 64 columns):
//   S_t = (scale*Q) @ K_t^T  (WMMA) -> online softmax -> O += P_t @ V_t (WMMA)
// Each wave owns a 16x32 O strip; A-fragments reused across both tiles.
// =====================================================================
__global__ __launch_bounds__(256)
void attn_kernel(const float* __restrict__ qkv, const int* __restrict__ ridx,
                 float* __restrict__ out) {
    __shared__ alignas(32) _Float16 Qs[W2 * HDQ];    // [qpix][d]
    __shared__ alignas(32) _Float16 Ks[W2 * HDQ];    // [kpix][d]
    __shared__ alignas(32) _Float16 Vst[HDQ * W2];   // transposed: [d][kpix]
    __shared__ alignas(32) _Float16 Ps[W2 * W2];     // [qpix][kpix] exp probs
    __shared__ float Ss[W2 * W2];                    // [qpix][kpix] scores
    __shared__ float rowmax[W2], rowsum[W2], rfac[W2];

    const int h = blockIdx.x, p = blockIdx.y, n = blockIdx.z;
    const int tid = threadIdx.x, lane = tid & 31, wave = tid >> 5;
    const int l15 = lane & 15, half = lane >> 4;
    const float scale = 0.044194173824159216f;

    const float* qb = qkv + ((size_t)n * OC + (size_t)h * HDQ) * HWPIX;
    const float* kb = qkv + ((size_t)n * OC + DIMC     + (size_t)h * HDQ) * HWPIX;
    const float* vb = qkv + ((size_t)n * OC + 2 * DIMC + (size_t)h * HDQ) * HWPIX;

    // stage Q (scaled) once: float4 over pixel quads (rows of 8 are contiguous)
    for (int i = tid; i < 1024; i += 256) {
        int d = i >> 4, q4 = (i & 15) << 2;
        v4f t = *(const v4f*)&qb[(size_t)d * HWPIX + win_pix(p, q4)];
#pragma unroll
        for (int j = 0; j < 4; ++j)
            Qs[(q4 + j) * HDQ + d] = (_Float16)(scale * t[j]);
    }
    if (tid < W2) { rowmax[tid] = -3.4e38f; rowsum[tid] = 0.f; }

    const int mt  = wave >> 1;        // M tile (shared by both N tiles)
    const int ntb = (wave & 1) * 2;   // base N tile
    v8f acco[2]; acco[0] = (v8f){}; acco[1] = (v8f){};

    for (int tch = 0; tch < TOPK; ++tch) {
        const int m = ridx[((size_t)(n * P2) + p) * TOPK + tch];
        __syncthreads();              // prev iter done with Ks/Vst/Ps (Qs staged)
        for (int i = tid; i < 1024; i += 256) {
            int d = i >> 4, k4 = (i & 15) << 2;
            int pix = win_pix(m, k4);
            v4f tk = *(const v4f*)&kb[(size_t)d * HWPIX + pix];
            v4f tv = *(const v4f*)&vb[(size_t)d * HWPIX + pix];
#pragma unroll
            for (int j = 0; j < 4; ++j)
                Ks[(k4 + j) * HDQ + d] = (_Float16)tk[j];
            v4h hv = { (_Float16)tv[0], (_Float16)tv[1], (_Float16)tv[2], (_Float16)tv[3] };
            *(v4h*)&Vst[d * W2 + k4] = hv;
        }
        __syncthreads();

        // ---- S = Q @ K^T : one A-fragment, two WMMAs per K-step ----
        {
            v8f s0 = {}, s1 = {};
#pragma unroll
            for (int k0 = 0; k0 < HDQ; k0 += 32) {
                v16h a  = frag_a(&Qs[(mt * 16 + l15) * HDQ + k0], half);
                v16h b0 = frag_b(&Ks[(ntb * 16 + l15) * HDQ + k0 + half * 16]);
                v16h b1 = frag_b(&Ks[((ntb + 1) * 16 + l15) * HDQ + k0 + half * 16]);
                s0 = __builtin_amdgcn_wmma_f32_16x16x32_f16(false, a, false, b0,
                                                            (short)0, s0, false, false);
                s1 = __builtin_amdgcn_wmma_f32_16x16x32_f16(false, a, false, b1,
                                                            (short)0, s1, false, false);
            }
#pragma unroll
            for (int v = 0; v < 8; ++v) {
                int r = (mt * 16 + half * 8 + v) * W2;
                Ss[r + ntb * 16 + l15]       = s0[v];
                Ss[r + (ntb + 1) * 16 + l15] = s1[v];
            }
        }
        __syncthreads();

        // ---- online softmax per row ----
        if (tid < W2) {
            float mo = rowmax[tid], mc = mo;
            for (int k = 0; k < W2; ++k) mc = fmaxf(mc, Ss[tid * W2 + k]);
            float f = __expf(mo - mc);
            float s = rowsum[tid] * f;
            for (int k = 0; k < W2; ++k) {
                float e = __expf(Ss[tid * W2 + k] - mc);
                s += e;
                Ps[tid * W2 + k] = (_Float16)e;
            }
            rowmax[tid] = mc; rowsum[tid] = s; rfac[tid] = f;
        }
        __syncthreads();

        // ---- rescale O, accumulate P @ V (A-fragment reused) ----
        {
#pragma unroll
            for (int i = 0; i < 2; ++i)
#pragma unroll
                for (int v = 0; v < 8; ++v)
                    acco[i][v] *= rfac[mt * 16 + half * 8 + v];
#pragma unroll
            for (int k0 = 0; k0 < W2; k0 += 32) {
                v16h a  = frag_a(&Ps[(mt * 16 + l15) * W2 + k0], half);
                v16h b0 = frag_b(&Vst[(ntb * 16 + l15) * W2 + k0 + half * 16]);
                v16h b1 = frag_b(&Vst[((ntb + 1) * 16 + l15) * W2 + k0 + half * 16]);
                acco[0] = __builtin_amdgcn_wmma_f32_16x16x32_f16(false, a, false, b0,
                                                                 (short)0, acco[0], false, false);
                acco[1] = __builtin_amdgcn_wmma_f32_16x16x32_f16(false, a, false, b1,
                                                                 (short)0, acco[1], false, false);
            }
        }
    }

    // ---- write O / rowsum ----
#pragma unroll
    for (int i = 0; i < 2; ++i) {
        int nt = ntb + i;
#pragma unroll
        for (int v = 0; v < 8; ++v) {
            int qp = mt * 16 + half * 8 + v;
            int d  = nt * 16 + l15;
            out[((size_t)n * DIMC + (size_t)(h * HDQ + d)) * HWPIX + win_pix(p, qp)] =
                acco[i][v] / rowsum[qp];
        }
    }
}

// =====================================================================
// Kernel 5: LePE depthwise 3x3 conv on V + bias, added into output.
// =====================================================================
__global__ void lepe_add_kernel(const float* __restrict__ qkv,
                                const float* __restrict__ wl, const float* __restrict__ bl,
                                float* __restrict__ out) {
    int idx = blockIdx.x * blockDim.x + threadIdx.x;
    if (idx >= BATCH * DIMC * HWPIX) return;
    int xw = idx % WW;
    int yh = (idx / WW) % HH;
    int c  = (idx / HWPIX) % DIMC;
    int n  = idx / (DIMC * HWPIX);
    const float* v = qkv + ((size_t)n * OC + 2 * DIMC + c) * HWPIX;
    float s = bl[c];
#pragma unroll
    for (int dy = 0; dy < 3; ++dy) {
#pragma unroll
        for (int dx = 0; dx < 3; ++dx) {
            int hy = yh + dy - 1, wx = xw + dx - 1;
            if (hy >= 0 && hy < HH && wx >= 0 && wx < WW)
                s += v[hy * WW + wx] * wl[c * 9 + dy * 3 + dx];
        }
    }
    out[idx] += s;
}

// =====================================================================
extern "C" void kernel_launch(void* const* d_in, const int* in_sizes, int n_in,
                              void* d_out, int out_size, void* d_ws, size_t ws_size,
                              hipStream_t stream) {
    const float* x      = (const float*)d_in[0];
    const float* w_qkv  = (const float*)d_in[1];
    const float* b_qkv  = (const float*)d_in[2];
    const float* w_lepe = (const float*)d_in[3];
    const float* b_lepe = (const float*)d_in[4];
    float* out = (float*)d_out;

    char* ws = (char*)d_ws;
    size_t off = 0;
    float* qkv = (float*)(ws + off); off += (size_t)BATCH * OC * HWPIX * sizeof(float);
    float* qw  = (float*)(ws + off); off += (size_t)BATCH * P2 * DIMC * sizeof(float);
    float* kw  = (float*)(ws + off); off += (size_t)BATCH * P2 * DIMC * sizeof(float);
    int*   rix = (int*)  (ws + off); off += (size_t)BATCH * P2 * TOPK * sizeof(int);
    (void)ws_size; (void)in_sizes; (void)n_in; (void)out_size;

    // 1) QKV projection GEMM (WMMA)
    dim3 g1(HWPIX / 32, OC / 128, BATCH);
    qkv_gemm_kernel<<<g1, 256, 0, stream>>>(x, w_qkv, b_qkv, qkv);

    // 2) window means
    int tot = BATCH * P2 * DIMC;
    winmean_kernel<<<(tot + 255) / 256, 256, 0, stream>>>(qkv, qw, kw);

    // 3) routing top-k
    routing_kernel<<<dim3(P2, BATCH), 64, 0, stream>>>(qw, kw, rix);

    // 4) fused flash attention over routed windows (WMMA)
    attn_kernel<<<dim3(HEADS, P2, BATCH), 256, 0, stream>>>(qkv, rix, out);

    // 5) LePE depthwise conv + bias added into output
    int tot2 = BATCH * DIMC * HWPIX;
    lepe_add_kernel<<<(tot2 + 255) / 256, 256, 0, stream>>>(qkv, w_lepe, b_lepe, out);
}
